// ChannelAttention_32478542692511
// MI455X (gfx1250) — compile-verified, matrix operand order checked by reading, not run on hardware
//
#include <hip/hip_runtime.h>
#include <hip/hip_bf16.h>

// ---------------------------------------------------------------------------
// ChannelAttention for MI455X (gfx1250, wave32, WMMA).
//
// B=16, idf=1024, cdf=256, L=256, M=ih*iw=4096.
// Pipeline: ctx transpose -> sourceC GEMM -> attn-logits GEMM
//           -> softmax(+transpose) -> out GEMM.
//
// Design points for MI455X:
//  * all GEMMs on v_wmma_f32_16x16x32_bf16 (f32 accumulate)
//  * every WMMA operand fragment is two contiguous 16-byte loads per lane
//    (global_load_b128): context is pre-transposed ([B][L][cdf], 4 MB),
//    and sourceC (32 MB bf16) is materialized in BOTH [B][M][L] and
//    [B][L][M] layouts. All of these stay resident in the 192 MB L2,
//    so tile re-reads never touch HBM.
//  * each wave owns a 16x128 output strip (8 tiles sharing one A fragment):
//    the 256 MB weightedContext stream is read only 2x from HBM.
//  * compulsory HBM traffic ~0.8 GB @ 23.3 TB/s -> ~35 us floor; bf16 WMMA
//    compute (~77 GFLOP) lands in the same range, so this is balanced.
// ---------------------------------------------------------------------------

typedef __attribute__((ext_vector_type(16))) __bf16 v16bf;
typedef __attribute__((ext_vector_type(8)))  __bf16 v8bf;
typedef __attribute__((ext_vector_type(8)))  float  v8f;

#define B_   16
#define IDF  1024
#define CDF  256
#define L_   256
#define M_   4096

// Per-lane K-chunk bases for the 16-bit 16x32 A layout (and its B mirror):
// lanes 0-15 : elems 0..7 -> K 0..7,  elems 8..15 -> K 16..23
// lanes 16-31: elems 0..7 -> K 8..15, elems 8..15 -> K 24..31
__device__ __forceinline__ void frag_bases(int lane, int& rc, int& hf, int& klo, int& khi) {
    rc  = lane & 15;
    hf  = (lane >> 4) & 1;
    klo = hf << 3;
    khi = 16 + (hf << 3);
}

// Convert 8 contiguous f32 (16B-aligned) into fragment elems [base..base+7].
__device__ __forceinline__ void cvt8(const float* __restrict__ p, v16bf& a, int base) {
    const float4 x = ((const float4*)p)[0];
    const float4 y = ((const float4*)p)[1];
    a[base + 0] = (__bf16)x.x; a[base + 1] = (__bf16)x.y;
    a[base + 2] = (__bf16)x.z; a[base + 3] = (__bf16)x.w;
    a[base + 4] = (__bf16)y.x; a[base + 5] = (__bf16)y.y;
    a[base + 6] = (__bf16)y.z; a[base + 7] = (__bf16)y.w;
}

// Two contiguous 16B bf16 chunks -> one 16-elem fragment (b128 loads).
__device__ __forceinline__ v16bf load_frag(const __bf16* __restrict__ p, int klo, int khi) {
    const v8bf lo = *(const v8bf*)(p + klo);
    const v8bf hi = *(const v8bf*)(p + khi);
    return __builtin_shufflevector(lo, hi, 0,1,2,3,4,5,6,7,8,9,10,11,12,13,14,15);
}

// ---------------------------------------------------------------------------
// Kernel 0: ctxT[b][l][c] = ctx[b][c][l].  32x32 tiles via padded LDS;
// coalesced on both sides. 16 * 8 * 8 = 1024 blocks of 256 threads.
// ---------------------------------------------------------------------------
__global__ __launch_bounds__(256)
void k_transpose_ctx(const float* __restrict__ ctx, float* __restrict__ ctxT)
{
    __shared__ float tile[32][33];
    const int tb = blockIdx.x;
    const int b  = tb >> 6;
    const int c0 = ((tb >> 3) & 7) << 5;
    const int l0 = (tb & 7) << 5;
    const int tx = threadIdx.x & 31;
    const int ty = threadIdx.x >> 5;            // 0..7

    const float* src = ctx + (size_t)b * CDF * L_;
    #pragma unroll
    for (int r = 0; r < 32; r += 8)
        tile[ty + r][tx] = src[(size_t)(c0 + ty + r) * L_ + l0 + tx];
    __syncthreads();
    float* dst = ctxT + (size_t)b * L_ * CDF;
    #pragma unroll
    for (int r = 0; r < 32; r += 8)
        dst[(size_t)(l0 + ty + r) * CDF + c0 + tx] = tile[tx][ty + r];
}

// ---------------------------------------------------------------------------
// Kernel 1: sourceC[b][m][l] = sum_c w2[m][c]*ctxT[b][l][c]; bf16 out in
// two layouts ([M][L] and [L][M]). 16 batches * 256 mTiles * 4 strips waves.
// ---------------------------------------------------------------------------
__global__ __launch_bounds__(256)
void k_sourceC(const float* __restrict__ w2, const float* __restrict__ ctxT,
               __bf16* __restrict__ srcC_ml, __bf16* __restrict__ srcC_lm)
{
    const int lane = threadIdx.x & 31;
    const int wave = blockIdx.x * 8 + (threadIdx.x >> 5);
    const int b  = wave >> 10;            // 1024 waves per batch
    const int t  = wave & 1023;
    const int m0 = (t >> 2) << 4;         // 256 m-tiles
    const int l0 = (t & 3) << 6;          // 4 strips of 64 columns

    int rc, hf, klo, khi;
    frag_bases(lane, rc, hf, klo, khi);

    const float* Arow  = w2 + (size_t)(m0 + rc) * CDF;
    const float* Bbase = ctxT + (size_t)b * L_ * CDF;

    v8f acc[4] = {};
    for (int k0 = 0; k0 < CDF; k0 += 32) {
        v16bf a;
        cvt8(Arow + k0 + klo, a, 0);
        cvt8(Arow + k0 + khi, a, 8);
        #pragma unroll
        for (int j = 0; j < 4; ++j) {
            const float* Brow = Bbase + (size_t)(l0 + 16 * j + rc) * CDF + k0;
            v16bf bb;
            cvt8(Brow + klo, bb, 0);
            cvt8(Brow + khi, bb, 8);
            acc[j] = __builtin_amdgcn_wmma_f32_16x16x32_bf16(
                false, a, false, bb, (short)0, acc[j], false, false);
        }
    }

    __bf16* out_ml = srcC_ml + (size_t)b * M_ * L_;
    __bf16* out_lm = srcC_lm + (size_t)b * L_ * M_;
    #pragma unroll
    for (int j = 0; j < 4; ++j) {
        const int col = l0 + 16 * j + rc;          // l
        #pragma unroll
        for (int r = 0; r < 8; ++r) {
            const int row = m0 + r + (hf << 3);    // m
            const __bf16 v = (__bf16)acc[j][r];
            out_ml[(size_t)row * L_ + col] = v;
            out_lm[(size_t)col * M_ + row] = v;
        }
    }
}

// ---------------------------------------------------------------------------
// Kernel 2: logits[b][i][l] = sum_m wC[b][i][m]*srcC[b][m][l]; K=4096.
// B operand from srcC_lm ([L][M]) -> per-lane contiguous b128 loads (L2-hot).
// Each wave: one i-tile x 8 l-tiles. 16*64*2 = 2048 waves.
// ---------------------------------------------------------------------------
__global__ __launch_bounds__(256)
void k_logits(const float* __restrict__ wC, const __bf16* __restrict__ srcC_lm,
              float* __restrict__ logit)
{
    const int lane = threadIdx.x & 31;
    const int wave = blockIdx.x * 8 + (threadIdx.x >> 5);
    const int b   = wave >> 7;            // 128 waves per batch
    const int t   = wave & 127;
    const int i0  = (t >> 1) << 4;        // 64 i-tiles
    const int l00 = (t & 1) << 7;         // 2 halves of 128 l-columns

    int rc, hf, klo, khi;
    frag_bases(lane, rc, hf, klo, khi);

    const float*  Arow = wC + ((size_t)b * IDF + i0 + rc) * M_;
    const __bf16* Bb   = srcC_lm + (size_t)b * L_ * M_;

    v8f acc[8] = {};
    for (int k0 = 0; k0 < M_; k0 += 32) {
        __builtin_prefetch(Arow + k0 + 256, 0, 1);   // global_prefetch_b8
        v16bf a;
        cvt8(Arow + k0 + klo, a, 0);
        cvt8(Arow + k0 + khi, a, 8);
        #pragma unroll
        for (int j = 0; j < 8; ++j) {
            const __bf16* Brow = Bb + (size_t)(l00 + 16 * j + rc) * M_ + k0;
            const v16bf bb = load_frag(Brow, klo, khi);
            acc[j] = __builtin_amdgcn_wmma_f32_16x16x32_bf16(
                false, a, false, bb, (short)0, acc[j], false, false);
        }
    }

    float* out = logit + (size_t)b * IDF * L_;
    #pragma unroll
    for (int j = 0; j < 8; ++j) {
        const int col = l00 + 16 * j + rc;
        #pragma unroll
        for (int r = 0; r < 8; ++r) {
            const int row = i0 + r + (hf << 3);
            out[(size_t)row * L_ + col] = acc[j][r];
        }
    }
}

// ---------------------------------------------------------------------------
// Kernel 3: softmax over L per (b,i) row; emit f32 attn_c [B][L][IDF] (output)
//           and bf16 attn [B][IDF][L] (A operand of final GEMM).
// One wave32 per row; shuffle reductions.
// ---------------------------------------------------------------------------
__global__ __launch_bounds__(256)
void k_softmax(const float* __restrict__ logit, __bf16* __restrict__ attnb,
               float* __restrict__ attn_out)
{
    const int lane = threadIdx.x & 31;
    const int row  = blockIdx.x * 8 + (threadIdx.x >> 5);  // [0, B*IDF)
    const float* p = logit + (size_t)row * L_;

    float v[8];
    float mx = -3.402823e38f;
    #pragma unroll
    for (int j = 0; j < 8; ++j) { v[j] = p[lane + 32 * j]; mx = fmaxf(mx, v[j]); }
    #pragma unroll
    for (int off = 16; off > 0; off >>= 1) mx = fmaxf(mx, __shfl_xor(mx, off, 32));

    float s = 0.0f;
    #pragma unroll
    for (int j = 0; j < 8; ++j) { v[j] = __expf(v[j] - mx); s += v[j]; }
    #pragma unroll
    for (int off = 16; off > 0; off >>= 1) s += __shfl_xor(s, off, 32);
    const float inv = 1.0f / s;

    const int b = row >> 10;     // / IDF
    const int i = row & 1023;
    __bf16* ab = attnb + (size_t)row * L_;
    #pragma unroll
    for (int j = 0; j < 8; ++j) {
        const int   l  = lane + 32 * j;
        const float pr = v[j] * inv;
        ab[l] = (__bf16)pr;
        attn_out[((size_t)b * L_ + l) * IDF + i] = pr;
    }
}

// ---------------------------------------------------------------------------
// Kernel 4: out[b][i][m] = sum_l attn[b][i][l]*srcC[b][m][l]; K=L=256.
// Both operands bf16 with K contiguous per lane (b128 loads, L2-hot).
// Each wave: one i-tile x 8 m-tiles. 16*64*32 = 32768 waves.
// ---------------------------------------------------------------------------
__global__ __launch_bounds__(256)
void k_out(const __bf16* __restrict__ attnb, const __bf16* __restrict__ srcC_ml,
           float* __restrict__ out)
{
    const int lane = threadIdx.x & 31;
    const int wave = blockIdx.x * 8 + (threadIdx.x >> 5);
    const int b   = wave >> 11;           // 2048 waves per batch
    const int t   = wave & 2047;
    const int i0  = (t >> 5) << 4;        // 64 i-tiles
    const int m00 = (t & 31) << 7;        // 32 strips of 128 m-columns

    int rc, hf, klo, khi;
    frag_bases(lane, rc, hf, klo, khi);

    const __bf16* Arow = attnb + ((size_t)b * IDF + i0 + rc) * L_;
    const __bf16* Bb   = srcC_ml + (size_t)b * M_ * L_;

    v8f acc[8] = {};
    for (int k0 = 0; k0 < L_; k0 += 32) {
        const v16bf a = load_frag(Arow + k0, klo, khi);
        #pragma unroll
        for (int j = 0; j < 8; ++j) {
            const __bf16* Brow = Bb + (size_t)(m00 + 16 * j + rc) * L_ + k0;
            const v16bf bb = load_frag(Brow, klo, khi);
            acc[j] = __builtin_amdgcn_wmma_f32_16x16x32_bf16(
                false, a, false, bb, (short)0, acc[j], false, false);
        }
    }

    float* outb = out + (size_t)b * IDF * M_;
    #pragma unroll
    for (int j = 0; j < 8; ++j) {
        const int col = m00 + 16 * j + rc;
        #pragma unroll
        for (int r = 0; r < 8; ++r) {
            const int row = i0 + r + (hf << 3);
            outb[(size_t)row * M_ + col] = acc[j][r];
        }
    }
}

// ---------------------------------------------------------------------------
extern "C" void kernel_launch(void* const* d_in, const int* in_sizes, int n_in,
                              void* d_out, int out_size, void* d_ws, size_t ws_size,
                              hipStream_t stream) {
    const float* wC  = (const float*)d_in[0];  // [B, idf, M]
    const float* ctx = (const float*)d_in[1];  // [B, cdf, L]
    const float* w2  = (const float*)d_in[2];  // [M, cdf]
    // d_in[3]=ih, d_in[4]=iw (compile-time constants here)

    float* out0 = (float*)d_out;                      // [B, idf, M]
    float* out1 = out0 + (size_t)B_ * IDF * M_;       // attn_c [B, L, idf]

    char* ws = (char*)d_ws;
    const size_t szC = (size_t)B_ * M_ * L_ * 2;      // 32 MB each
    __bf16* srcC_ml = (__bf16*)ws;
    __bf16* srcC_lm = (__bf16*)(ws + szC);
    float*  logit   = (float*)(ws + 2 * szC);                               // 16 MB
    __bf16* attnb   = (__bf16*)(ws + 2 * szC + (size_t)B_ * IDF * L_ * 4);  //  8 MB
    float*  ctxT    = (float*)(ws + 2 * szC + (size_t)B_ * IDF * L_ * 4
                                  + (size_t)B_ * IDF * L_ * 2);             //  4 MB

    k_transpose_ctx<<<1024, 256, 0, stream>>>(ctx, ctxT);
    k_sourceC<<<2048, 256, 0, stream>>>(w2, ctxT, srcC_ml, srcC_lm); // 16384 waves
    k_logits <<<256,  256, 0, stream>>>(wC, srcC_lm, logit);         //  2048 waves
    k_softmax<<<2048, 256, 0, stream>>>(logit, attnb, out1);         // 16384 rows
    k_out    <<<4096, 256, 0, stream>>>(attnb, srcC_ml, out0);       // 32768 waves
}
